// LSTM_25950192403294
// MI455X (gfx1250) — matrix operand order repaired and checked by reading, not compiled, not run on hardware
//
#include <hip/hip_runtime.h>
#include <hip/hip_bf16.h>
#include <math.h>
#include <stdint.h>

#define EMBD 256
#define HIDD 1024
#define NTRK 4096
#define SEQL 9
#define NPRED 12
#define TSTEPS (SEQL - 1 + NPRED - 1)   // 19 output steps
#define SCALE_F 4.0f
#define KC 64                            // K-chunk staged per TDM transfer
#define NCH (EMBD / KC + HIDD / KC)      // 4 + 16 = 20 chunks

typedef __bf16 bhalf;
typedef __attribute__((ext_vector_type(16))) __bf16 v16bf;
typedef __attribute__((ext_vector_type(8)))  __bf16 v8bf;
typedef __attribute__((ext_vector_type(8)))  float  v8f;

#ifndef __has_builtin
#define __has_builtin(x) 0
#endif
#if __has_builtin(__builtin_amdgcn_tensor_load_to_lds) && \
    __has_builtin(__builtin_amdgcn_s_wait_tensorcnt)
#define HAS_TDM 1
#else
#define HAS_TDM 0
#endif

__device__ __forceinline__ float sigmoidf_(float x) {
    return 1.0f / (1.0f + __expf(-x));
}

// A-fragment for v_wmma_f32_16x16x32_bf16 from a row-major row:
// lane l (m = l&15) holds K = kbase..kbase+15
// (kbase = (l<16)?0:16): two contiguous 16B chunks.
__device__ __forceinline__ v16bf load_a16(const bhalf* p) {
    union { v16bf v; v8bf h[2]; } u;
    u.h[0] = *(const v8bf*)(p);
    u.h[1] = *(const v8bf*)(p + 8);
    return u.v;
}

#if HAS_TDM
// ---------------------------------------------------------------------------
// TDM: DMA a 3D weight panel  (KC contiguous K) x (32 rows) x (4 gate groups)
// into LDS, per ISA D# layout (CDNA5 ISA).
//   addr = global_addr + 2*( x + y*tensor_dim0_stride + z*tensor_dim1_stride )
//   x in [0,KC), y in [0,32) rows (stride ld), z in [0,4) gates (stride HID*ld)
// LDS dest is tile-contiguous: ((z*32+y)*KC + x)*2 bytes.
// ---------------------------------------------------------------------------
typedef __attribute__((ext_vector_type(4))) unsigned int v4u_;
typedef __attribute__((ext_vector_type(8))) int v8i_;
typedef __attribute__((ext_vector_type(4))) int v4i_;

__device__ __forceinline__ void tdm_load_b_panel(const bhalf* wbase, int ld,
                                                 int j0, int k0, bhalf* ldsdst) {
    unsigned long long ga = (unsigned long long)(uintptr_t)wbase +
                            2ull * ((unsigned long long)j0 * (unsigned)ld +
                                    (unsigned long long)k0);
    unsigned int ldsa = (unsigned int)(uintptr_t)ldsdst;   // LDS byte offset
    unsigned long long s0 = (unsigned long long)(unsigned)ld;          // dim0 stride
    unsigned long long s1 = (unsigned long long)HIDD * (unsigned)ld;   // dim1 stride

    v4u_ g0;
    g0.x = 1u;                                             // count=1, user D#
    g0.y = ldsa;                                           // lds_addr
    g0.z = (unsigned int)(ga & 0xFFFFFFFFu);               // global_addr[31:0]
    g0.w = (unsigned int)((ga >> 32) & 0x01FFFFFFu) | (2u << 30); // addr[56:32]|type=2

    v8i_ g1;
    g1[0] = (int)(1u << 16);                               // data_size = 2 bytes
    g1[1] = (int)(((unsigned)ld & 0xFFFFu) << 16);         // tensor_dim0[15:0]
    g1[2] = (int)((((unsigned)ld >> 16) & 0xFFFFu) |
                  (((unsigned)HIDD & 0xFFFFu) << 16));     // dim0 hi | tensor_dim1 lo
    g1[3] = (int)((((unsigned)HIDD >> 16) & 0xFFFFu) |
                  ((unsigned)KC << 16));                   // dim1 hi | tile_dim0
    g1[4] = (int)(32u | (4u << 16));                       // tile_dim1=32, tile_dim2=4
    g1[5] = (int)(unsigned)(s0 & 0xFFFFFFFFull);           // dim0_stride[31:0]
    g1[6] = (int)((unsigned)((s0 >> 32) & 0xFFFFull) |
                  ((unsigned)(s1 & 0xFFFFull) << 16));     // dim0_str hi | dim1_str lo
    g1[7] = (int)(unsigned)((s1 >> 16) & 0xFFFFFFFFull);   // dim1_stride[47:16]

    v4i_ g2 = {4, 0, 0, 0};                                // tensor_dim2=4 (gates)
    v4i_ g3 = {0, 0, 0, 0};
#if defined(__clang_major__) && __clang_major__ >= 23
    v8i_ g4 = {0, 0, 0, 0, 0, 0, 0, 0};
    __builtin_amdgcn_tensor_load_to_lds(g0, g1, g2, g3, g4, 0);
#else
    __builtin_amdgcn_tensor_load_to_lds(g0, g1, g2, g3, 0);
#endif
}
#endif  // HAS_TDM

// ---------------------------------------------------------------------------
// fp32 -> bf16 weight conversion
// ---------------------------------------------------------------------------
__global__ __launch_bounds__(256) void cvt_bf16_kernel(const float* __restrict__ src,
                                                       bhalf* __restrict__ dst, int n) {
    int i = blockIdx.x * 256 + threadIdx.x;
    if (i < n) dst[i] = (bhalf)src[i];
}

// ---------------------------------------------------------------------------
// zero h, c (f32) and h (bf16)
// ---------------------------------------------------------------------------
__global__ __launch_bounds__(256) void init_state_kernel(float* __restrict__ H,
                                                         float* __restrict__ C,
                                                         bhalf* __restrict__ Hb) {
    size_t i = (size_t)blockIdx.x * 256 + threadIdx.x;
    if (i < (size_t)NTRK * HIDD) {
        H[i] = 0.0f;
        C[i] = 0.0f;
        Hb[i] = (bhalf)0.0f;
    }
}

// ---------------------------------------------------------------------------
// Embed: x = relu(SCALE*(o2-o1) @ emb_w^T + emb_b), last 2 cols zero.
// tagidx >= 0 -> one-hot tag input, mask forced to 1.
// ---------------------------------------------------------------------------
__global__ __launch_bounds__(256) void embed_kernel(const float* __restrict__ o1,
                                                    const float* __restrict__ o2,
                                                    const float* __restrict__ emb_w,
                                                    const float* __restrict__ emb_b,
                                                    bhalf* __restrict__ X,
                                                    int* __restrict__ Mask,
                                                    int tagidx) {
    const int track = blockIdx.x;
    const int e = threadIdx.x;
    if (tagidx >= 0) {
        X[(size_t)track * EMBD + e] = (bhalf)((e == tagidx) ? 1.0f : 0.0f);
        if (e == 0) Mask[track] = 1;
        return;
    }
    const float o1x = o1[(size_t)track * 2 + 0];
    const float o1y = o1[(size_t)track * 2 + 1];
    const float o2x = o2[(size_t)track * 2 + 0];
    const float o2y = o2[(size_t)track * 2 + 1];
    const float v0 = o2x - o1x;
    const float v1 = o2y - o1y;
    float x = 0.0f;
    if (e < EMBD - 2) {
        float r = SCALE_F * (v0 * emb_w[2 * e] + v1 * emb_w[2 * e + 1]) + emb_b[e];
        x = fmaxf(r, 0.0f);
    }
    X[(size_t)track * EMBD + e] = (bhalf)x;
    if (e == 0) Mask[track] = !((o1x != o1x) || (o2x != o2x));
}

// ---------------------------------------------------------------------------
// Fused LSTM cell. Grid (32, 32); block 256 = 8 waves.
// Block: 128 tracks x 32 hidden units x 4 gates.
// Wave (mtile = w&3, jsub = w>>2): 2 M-tiles x 16 units x 4 gates
//   -> 8 x v_wmma_f32_16x16x32_bf16 per 32-K step.
// B (weight) panel double-buffered in LDS via TDM (TENSORcnt) or coop copy.
// ---------------------------------------------------------------------------
__global__ __launch_bounds__(256) void lstm_cell_wmma(const bhalf* __restrict__ X,
                                                      const bhalf* __restrict__ Hin,
                                                      bhalf* __restrict__ Hout,
                                                      const bhalf* __restrict__ Wih,
                                                      const bhalf* __restrict__ Whh,
                                                      const float* __restrict__ Bias,
                                                      const int* __restrict__ Mask,
                                                      float* __restrict__ Hst,
                                                      float* __restrict__ Cst) {
    __shared__ alignas(64) bhalf sB[2][4 * 32 * KC];   // 2 x 16KB

    const int tid   = threadIdx.x;
    const int wv    = tid >> 5;
    const int lane  = tid & 31;
    const int mtile = wv & 3;
    const int jsub  = wv >> 2;
    const int nlane = lane & 15;
    const int hs    = lane >> 4;                     // half-wave select
    const int m0    = blockIdx.x * 128 + mtile * 32; // 2 M-tiles per wave
    const int j0    = blockIdx.y * 32;               // block j-range
    const int jloc  = jsub * 16 + nlane;
    const int j     = j0 + jloc;
    const int ar0   = m0 + nlane;                    // A rows, tile 0
    const int ar1   = m0 + 16 + nlane;               // A rows, tile 1

    v8f acc0[4] = {};   // gates for M-tile 0
    v8f acc1[4] = {};   // gates for M-tile 1

#if HAS_TDM
    if (wv == 0) tdm_load_b_panel(Wih, EMBD, j0, 0, &sB[0][0]);
#endif

    for (int c = 0; c < NCH; ++c) {
#if HAS_TDM
        if (wv == 0) {
            const int nc = c + 1;
            if (nc < NCH) {
                if (nc < EMBD / KC)
                    tdm_load_b_panel(Wih, EMBD, j0, nc * KC, &sB[nc & 1][0]);
                else
                    tdm_load_b_panel(Whh, HIDD, j0, (nc - EMBD / KC) * KC,
                                     &sB[nc & 1][0]);
                __builtin_amdgcn_s_wait_tensorcnt((short)1); // chunk c landed
            } else {
                __builtin_amdgcn_s_wait_tensorcnt((short)0);
            }
        }
#else
        {   // cooperative copy fallback: 128 rows x KC elements -> sB[c&1]
            const bhalf* wb; int ld, k0;
            if (c < EMBD / KC) { wb = Wih; ld = EMBD; k0 = c * KC; }
            else               { wb = Whh; ld = HIDD; k0 = (c - EMBD / KC) * KC; }
            bhalf* dst = &sB[c & 1][0];
            for (int e = tid * 8; e < 4 * 32 * KC; e += 256 * 8) {
                const int x   = e & (KC - 1);
                const int row = e / KC;            // 0..127
                const int g   = row >> 5, y = row & 31;
                const bhalf* src =
                    wb + ((size_t)(g * HIDD + j0 + y) * ld + k0 + x);
                *(v8bf*)(dst + e) = *(const v8bf*)src;
            }
        }
#endif
        __syncthreads();

        const bhalf* bb = &sB[c & 1][0];
        const bhalf* a0p;
        const bhalf* a1p;
        if (c < EMBD / KC) {
            a0p = X + (size_t)ar0 * EMBD + c * KC + hs * 16;
            a1p = X + (size_t)ar1 * EMBD + c * KC + hs * 16;
        } else {
            a0p = Hin + (size_t)ar0 * HIDD + (c - EMBD / KC) * KC + hs * 16;
            a1p = Hin + (size_t)ar1 * HIDD + (c - EMBD / KC) * KC + hs * 16;
        }
        #pragma unroll
        for (int kk = 0; kk < KC; kk += 32) {
            v16bf a0 = load_a16(a0p + kk);
            v16bf a1 = load_a16(a1p + kk);
            #pragma unroll
            for (int g = 0; g < 4; ++g) {
                v16bf b = *(const v16bf*)(bb + (size_t)((g * 32 + jloc) * KC + kk) +
                                          hs * 16);
                acc0[g] = __builtin_amdgcn_wmma_f32_16x16x32_bf16(
                    false, a0, false, b, (short)0, acc0[g], false, false);
                acc1[g] = __builtin_amdgcn_wmma_f32_16x16x32_bf16(
                    false, a1, false, b, (short)0, acc1[g], false, false);
            }
        }
        __syncthreads();
    }

    // ---- epilogue: gates + masked state update, directly on D tiles ----
    const float bi = Bias[0 * HIDD + j];
    const float bf = Bias[1 * HIDD + j];
    const float bg = Bias[2 * HIDD + j];
    const float bo = Bias[3 * HIDD + j];
    #pragma unroll
    for (int mt = 0; mt < 2; ++mt) {
        const v8f* acc = mt ? acc1 : acc0;
        const int mbase = m0 + mt * 16 + hs * 8;   // vgpr r -> track mbase+r
        #pragma unroll
        for (int r = 0; r < 8; ++r) {
            const int t = mbase + r;
            const int msk = Mask[t];
            const float iv = sigmoidf_(acc[0][r] + bi);
            const float fv = sigmoidf_(acc[1][r] + bf);
            const float gv = tanhf(acc[2][r] + bg);
            const float ov = sigmoidf_(acc[3][r] + bo);
            const size_t idx = (size_t)t * HIDD + j;
            const float oc = Cst[idx];
            const float oh = Hst[idx];
            const float c2 = fv * oc + iv * gv;
            const float h2 = ov * tanhf(c2);
            const float cf = msk ? c2 : oc;
            const float hf = msk ? h2 : oh;
            Cst[idx] = cf;
            Hst[idx] = hf;
            Hout[idx] = (bhalf)hf;
        }
    }
}

// ---------------------------------------------------------------------------
// h2n head + output write + position carry update (one wave per track).
// ---------------------------------------------------------------------------
__global__ __launch_bounds__(256) void h2n_kernel(const float* __restrict__ H,
                                                  const int* __restrict__ Mask,
                                                  const float* __restrict__ Wn,
                                                  const float* __restrict__ Bn,
                                                  const float* __restrict__ obs2,
                                                  float* __restrict__ P1,
                                                  float* __restrict__ P2,
                                                  float* __restrict__ outN,
                                                  float* __restrict__ outP) {
    const int wv = threadIdx.x >> 5;
    const int lane = threadIdx.x & 31;
    const int track = blockIdx.x * 8 + wv;
    const float* h = H + (size_t)track * HIDD;
    float r0 = 0.f, r1 = 0.f, r2 = 0.f, r3 = 0.f, r4 = 0.f;
    for (int jj = lane; jj < HIDD; jj += 32) {
        const float hv = h[jj];
        r0 += hv * Wn[0 * HIDD + jj];
        r1 += hv * Wn[1 * HIDD + jj];
        r2 += hv * Wn[2 * HIDD + jj];
        r3 += hv * Wn[3 * HIDD + jj];
        r4 += hv * Wn[4 * HIDD + jj];
    }
    #pragma unroll
    for (int o = 16; o > 0; o >>= 1) {
        r0 += __shfl_xor(r0, o, 32);
        r1 += __shfl_xor(r1, o, 32);
        r2 += __shfl_xor(r2, o, 32);
        r3 += __shfl_xor(r3, o, 32);
        r4 += __shfl_xor(r4, o, 32);
    }
    if (lane == 0) {
        r0 += Bn[0]; r1 += Bn[1]; r2 += Bn[2]; r3 += Bn[3]; r4 += Bn[4];
        const float s2 = 0.01f + 0.2f * sigmoidf_(r2);
        const float s3 = 0.01f + 0.2f * sigmoidf_(r3);
        const float rh = 0.7f * sigmoidf_(r4);
        const float qn = __builtin_nanf("");
        const int msk = Mask[track];
        const float n0 = msk ? r0 : qn;
        const float n1 = msk ? r1 : qn;
        outN[(size_t)track * 5 + 0] = n0;
        outN[(size_t)track * 5 + 1] = n1;
        outN[(size_t)track * 5 + 2] = msk ? s2 : qn;
        outN[(size_t)track * 5 + 3] = msk ? s3 : qn;
        outN[(size_t)track * 5 + 4] = msk ? rh : qn;
        const float o2x = obs2[(size_t)track * 2 + 0];
        const float o2y = obs2[(size_t)track * 2 + 1];
        const float px = o2x + n0;
        const float py = o2y + n1;
        outP[(size_t)track * 2 + 0] = px;
        outP[(size_t)track * 2 + 1] = py;
        const float op2x = P2[(size_t)track * 2 + 0];
        const float op2y = P2[(size_t)track * 2 + 1];
        P1[(size_t)track * 2 + 0] = op2x;
        P1[(size_t)track * 2 + 1] = op2y;
        P2[(size_t)track * 2 + 0] = px;
        P2[(size_t)track * 2 + 1] = py;
    }
}

// ---------------------------------------------------------------------------
// host-side orchestration
// ---------------------------------------------------------------------------
extern "C" void kernel_launch(void* const* d_in, const int* in_sizes, int n_in,
                              void* d_out, int out_size, void* d_ws, size_t ws_size,
                              hipStream_t stream) {
    (void)in_sizes; (void)n_in; (void)out_size; (void)ws_size;
    const float* observed = (const float*)d_in[0];
    const float* emb_w    = (const float*)d_in[1];
    const float* emb_b    = (const float*)d_in[2];
    const float* enc_wih  = (const float*)d_in[3];
    const float* enc_whh  = (const float*)d_in[4];
    const float* enc_b    = (const float*)d_in[5];
    const float* dec_wih  = (const float*)d_in[6];
    const float* dec_whh  = (const float*)d_in[7];
    const float* dec_b    = (const float*)d_in[8];
    const float* h2n_w    = (const float*)d_in[9];
    const float* h2n_b    = (const float*)d_in[10];
    float* out = (float*)d_out;

    char* ws = (char*)d_ws;
    size_t off = 0;
    auto carve = [&](size_t bytes) -> char* {
        char* p = ws + off;
        off = (off + bytes + 255) & ~(size_t)255;
        return p;
    };
    const int NIH = 4 * HIDD * EMBD;
    const int NHH = 4 * HIDD * HIDD;
    bhalf* WihE = (bhalf*)carve((size_t)NIH * 2);
    bhalf* WhhE = (bhalf*)carve((size_t)NHH * 2);
    bhalf* WihD = (bhalf*)carve((size_t)NIH * 2);
    bhalf* WhhD = (bhalf*)carve((size_t)NHH * 2);
    bhalf* Xb   = (bhalf*)carve((size_t)NTRK * EMBD * 2);
    bhalf* Hb0  = (bhalf*)carve((size_t)NTRK * HIDD * 2);
    bhalf* Hb1  = (bhalf*)carve((size_t)NTRK * HIDD * 2);
    float* Hf   = (float*)carve((size_t)NTRK * HIDD * 4);
    float* Cf   = (float*)carve((size_t)NTRK * HIDD * 4);
    int*   Msk  = (int*)  carve((size_t)NTRK * 4);
    float* P1   = (float*)carve((size_t)NTRK * 2 * 4);
    float* P2   = (float*)carve((size_t)NTRK * 2 * 4);

    cvt_bf16_kernel<<<(NIH + 255) / 256, 256, 0, stream>>>(enc_wih, WihE, NIH);
    cvt_bf16_kernel<<<(NHH + 255) / 256, 256, 0, stream>>>(enc_whh, WhhE, NHH);
    cvt_bf16_kernel<<<(NIH + 255) / 256, 256, 0, stream>>>(dec_wih, WihD, NIH);
    cvt_bf16_kernel<<<(NHH + 255) / 256, 256, 0, stream>>>(dec_whh, WhhD, NHH);
    init_state_kernel<<<(NTRK * HIDD + 255) / 256, 256, 0, stream>>>(Hf, Cf, Hb0);

    bhalf* HbIn = Hb0;
    bhalf* HbOut = Hb1;
    const dim3 cgrid(NTRK / 128, HIDD / 32);   // (32, 32)

    auto run_cell = [&](const bhalf* wih, const bhalf* whh, const float* bias) {
        lstm_cell_wmma<<<cgrid, 256, 0, stream>>>(Xb, HbIn, HbOut, wih, whh, bias,
                                                  Msk, Hf, Cf);
        bhalf* t = HbIn; HbIn = HbOut; HbOut = t;
    };
    auto run_h2n = [&](const float* obs2, int s) {
        h2n_kernel<<<NTRK / 8, 256, 0, stream>>>(
            Hf, Msk, h2n_w, h2n_b, obs2, P1, P2,
            out + (size_t)s * NTRK * 5,
            out + (size_t)TSTEPS * NTRK * 5 + (size_t)s * NTRK * 2);
    };

    // encoder tag step
    embed_kernel<<<NTRK, 256, 0, stream>>>(nullptr, nullptr, emb_w, emb_b, Xb, Msk,
                                           EMBD - 2);
    run_cell(WihE, WhhE, enc_b);

    // encoder scan: t = 0..7
    for (int t = 0; t < SEQL - 1; ++t) {
        const float* o1 = observed + (size_t)t * NTRK * 2;
        const float* o2 = observed + (size_t)(t + 1) * NTRK * 2;
        embed_kernel<<<NTRK, 256, 0, stream>>>(o1, o2, emb_w, emb_b, Xb, Msk, -1);
        run_cell(WihE, WhhE, enc_b);
        run_h2n(o2, t);
    }

    // decoder tag step
    embed_kernel<<<NTRK, 256, 0, stream>>>(nullptr, nullptr, emb_w, emb_b, Xb, Msk,
                                           EMBD - 1);
    run_cell(WihD, WhhD, dec_b);

    // decoder scan: d = 0..10
    for (int d = 0; d < NPRED - 1; ++d) {
        embed_kernel<<<NTRK, 256, 0, stream>>>(P1, P2, emb_w, emb_b, Xb, Msk, -1);
        run_cell(WihD, WhhD, dec_b);
        run_h2n(P2, SEQL - 1 + d);
    }
}
